// ConditioningMoEINR_14104672600556
// MI455X (gfx1250) — matrix operand order repaired and checked by reading, not compiled, not run on hardware
//
#include <hip/hip_runtime.h>

typedef __attribute__((ext_vector_type(16))) _Float16 v16h;
typedef __attribute__((ext_vector_type(8)))  float    v8f;

union FragH { v16h v; uint4 q[2]; };

#define N_PTS        131072
#define ROWS_PER_BLK 64
#define THREADS      128

// Workspace offsets (in halfs). Each "frag" = 32 lanes x 16 halfs = 512 halfs.
#define OFF_ENC1 0              // 2*8 frags
#define OFF_ENC2 8192           // 4*8
#define OFF_POL0 24576          // 1*4
#define OFF_POL1 26624          // 2*4
#define OFF_POL2 30720          // 2*4
#define OFF_EW0  34816          // 7 * 6*8
#define OFF_EW1  206848         // 7 * 4*8
#define OFF_EW2  321536         // 7 * 4*8
#define EW0_PER  (6*8*512)
#define EW12_PER (4*8*512)

// ---------------- prep: f32 -> f16 B-fragment swizzle ----------------
// B frag layout (16-bit B, 32x16 per k-chunk): lane l -> col n = nt*16 + (l&15),
// k-group = (l>>4)*16, 16 consecutive K values per lane (element j -> K = kgrp + j).
__global__ void swizzle_w(const float* __restrict__ src, _Float16* __restrict__ dst,
                          int Kreal, int Ncols, int KC, int NT, int nmat)
{
    int  per   = KC * NT * 512;
    long total = (long)nmat * per;
    for (long i = blockIdx.x * (long)blockDim.x + threadIdx.x; i < total;
         i += (long)gridDim.x * blockDim.x) {
        int m    = (int)(i / per);
        int rrem = (int)(i % per);
        int frag = rrem >> 9;
        int w    = rrem & 511;
        int lane = w >> 4, h = w & 15;
        int kc   = frag / NT, nt = frag % NT;
        int k    = kc * 32 + (lane >> 4) * 16 + h;
        int n    = nt * 16 + (lane & 15);
        float v  = 0.0f;
        if (k < Kreal) v = src[((long)m * Kreal + k) * Ncols + n];
        dst[i] = (_Float16)v;
    }
}

// ---------------- generic WMMA layer: sOut = sin(30*(sIn @ W + b)) ----------------
// A frag (16-bit A, 16x32): lane l: m=l&15, g=l>>4; 16 contiguous halfs at k0+g*16.
// C/D (f32 16x16): elem r of lane l -> row = (l>>4)*8 + r, col = l&15.
template<int KC, int NT>
__device__ __forceinline__ void layer_wmma(
    const _Float16* __restrict__ sIn, int strideIn,
    const _Float16* __restrict__ wfrag,        // pre-swizzled f16 fragments in global
    const float*    __restrict__ bias,
    _Float16*       __restrict__ sOut, int strideOut,
    int lane, int rowBase)
{
    const int m = lane & 15, g = lane >> 4, nsub = lane & 15;
    FragH a[KC];
    const _Float16* arow = sIn + (rowBase + m) * strideIn + g * 16;
#pragma unroll
    for (int kc = 0; kc < KC; ++kc) {
        a[kc].q[0] = *(const uint4*)(arow + kc * 32);
        a[kc].q[1] = *(const uint4*)(arow + kc * 32 + 8);
    }
#pragma unroll
    for (int nt = 0; nt < NT; ++nt) {
        v8f c = {};
#pragma unroll
        for (int kc = 0; kc < KC; ++kc) {
            FragH b;
            const _Float16* bp = wfrag + (((long)(kc * NT + nt)) * 32 + lane) * 16;
            b.q[0] = *(const uint4*)(bp);
            b.q[1] = *(const uint4*)(bp + 8);
            c = __builtin_amdgcn_wmma_f32_16x16x32_f16(false, a[kc].v, false, b.v,
                                                       (short)0, c, false, false);
        }
        const float bv     = bias[nt * 16 + nsub];
        const int   colOut = nt * 16 + nsub;
#pragma unroll
        for (int r = 0; r < 8; ++r) {
            float v = __sinf(30.0f * (c[r] + bv));
            int rowOut = rowBase + g * 8 + r;
            sOut[rowOut * strideOut + colOut] = (_Float16)v;
        }
    }
}

// ---------------- main fused MoE-INR kernel ----------------
__global__ __launch_bounds__(THREADS) void moe_inr_main(
    const float* __restrict__ x,
    const _Float16* __restrict__ wsf,
    const float* __restrict__ enc_b1, const float* __restrict__ enc_b2,
    const float* __restrict__ pol_b0, const float* __restrict__ pol_b1,
    const float* __restrict__ pol_b2,
    const float* __restrict__ pol_Wl, const float* __restrict__ pol_bl,
    const float* __restrict__ exp_b0, const float* __restrict__ exp_b1,
    const float* __restrict__ exp_b2,
    const float* __restrict__ exp_Wo, const float* __restrict__ exp_bo,
    float* __restrict__ out)
{
    __shared__ __align__(16) _Float16 sFused[64 * 200];  // fused feats, K=192 live
    __shared__ __align__(16) _Float16 sA[64 * 136];
    __shared__ __align__(16) _Float16 sB[64 * 136];

    const int tid     = threadIdx.x;
    const int lane    = tid & 31;
    const int rowBase = (tid >> 5) * 16;
    const int blkRow0 = blockIdx.x * ROWS_PER_BLK;

    // ---- Positional encoding -> sA[:, 0:64] (cols 52..63 zero) ----
    for (int i = tid; i < 64 * 64; i += THREADS) {
        int r = i >> 6, cidx = i & 63;
        float v = 0.0f;
        if (cidx < 4) {
            v = x[(blkRow0 + r) * 4 + cidx];
        } else if (cidx < 52) {
            bool is_cos = cidx >= 28;
            int  j      = cidx - (is_cos ? 28 : 4);
            int  feat   = j / 6, fq = j % 6;
            float ang = x[(blkRow0 + r) * 4 + feat] * ((float)(1 << fq) * 3.14159265358979323846f);
            v = is_cos ? __cosf(ang) : __sinf(ang);
        }
        sA[r * 136 + cidx] = (_Float16)v;
    }
    __syncthreads();

    // ---- Encoder ----
    layer_wmma<2, 8>(sA, 136, wsf + OFF_ENC1, enc_b1, sB, 136, lane, rowBase);
    __syncthreads();
    layer_wmma<4, 8>(sB, 136, wsf + OFF_ENC2, enc_b2, sFused, 200, lane, rowBase);  // fused[:,0:128]
    __syncthreads();

    // ---- Policy input [x, indicator=0] padded to 32 -> sA ----
    for (int i = tid; i < 64 * 32; i += THREADS) {
        int r = i >> 5, cidx = i & 31;
        float v = (cidx < 4) ? x[(blkRow0 + r) * 4 + cidx] : 0.0f;
        sA[r * 136 + cidx] = (_Float16)v;
    }
    __syncthreads();
    layer_wmma<1, 4>(sA, 136, wsf + OFF_POL0, pol_b0, sB, 136, lane, rowBase);
    __syncthreads();
    layer_wmma<2, 4>(sB, 136, wsf + OFF_POL1, pol_b1, sA, 136, lane, rowBase);
    __syncthreads();
    layer_wmma<2, 4>(sA, 136, wsf + OFF_POL2, pol_b2, sFused + 128, 200, lane, rowBase); // fused[:,128:192]
    __syncthreads();

    // ---- Routing: softmax cancels -> w0 = 1/(1+exp(l1-l0)); keep in registers ----
    float wsel[7];
    float outacc = 0.0f;
    if (tid < 64) {
        float lg[7];
#pragma unroll
        for (int e = 0; e < 7; ++e) lg[e] = pol_bl[e];
        for (int k = 0; k < 64; ++k) {
            float av = (float)sFused[tid * 200 + 128 + k];
#pragma unroll
            for (int e = 0; e < 7; ++e) lg[e] += av * pol_Wl[k * 7 + e];
        }
        int i0 = 0; float v0 = lg[0];
#pragma unroll
        for (int e = 1; e < 7; ++e) if (lg[e] > v0) { v0 = lg[e]; i0 = e; }
        int i1 = -1; float v1 = -1e30f;
#pragma unroll
        for (int e = 0; e < 7; ++e) if (e != i0 && lg[e] > v1) { v1 = lg[e]; i1 = e; }
        float w0 = 1.0f / (1.0f + __expf(v1 - v0));
#pragma unroll
        for (int e = 0; e < 7; ++e) wsel[e] = 0.0f;
        wsel[i0] = w0;
        wsel[i1] = 1.0f - w0;
    }

    // ---- Experts (dense; indicator/pad K-rows dropped -> K=192) ----
    for (int e = 0; e < 7; ++e) {
        layer_wmma<6, 8>(sFused, 200, wsf + OFF_EW0 + (long)e * EW0_PER,  exp_b0 + e * 128, sB, 136, lane, rowBase);
        __syncthreads();
        layer_wmma<4, 8>(sB, 136,    wsf + OFF_EW1 + (long)e * EW12_PER, exp_b1 + e * 128, sA, 136, lane, rowBase);
        __syncthreads();
        layer_wmma<4, 8>(sA, 136,    wsf + OFF_EW2 + (long)e * EW12_PER, exp_b2 + e * 128, sB, 136, lane, rowBase);
        __syncthreads();
        if (tid < 64) {
            float acc = exp_bo[e];
            for (int k = 0; k < 128; ++k)
                acc += (float)sB[tid * 136 + k] * exp_Wo[e * 128 + k];
            outacc += wsel[e] * acc;
        }
        __syncthreads();
    }

    if (tid < 64) out[blkRow0 + tid] = outacc;
}

// ---------------- host entry ----------------
extern "C" void kernel_launch(void* const* d_in, const int* in_sizes, int n_in,
                              void* d_out, int out_size, void* d_ws, size_t ws_size,
                              hipStream_t stream)
{
    const float* x      = (const float*)d_in[0];
    const float* enc_W1 = (const float*)d_in[1];
    const float* enc_b1 = (const float*)d_in[2];
    const float* enc_W2 = (const float*)d_in[3];
    const float* enc_b2 = (const float*)d_in[4];
    const float* pol_W0 = (const float*)d_in[5];
    const float* pol_b0 = (const float*)d_in[6];
    const float* pol_W1 = (const float*)d_in[7];
    const float* pol_b1 = (const float*)d_in[8];
    const float* pol_W2 = (const float*)d_in[9];
    const float* pol_b2 = (const float*)d_in[10];
    const float* pol_Wl = (const float*)d_in[11];
    const float* pol_bl = (const float*)d_in[12];
    const float* exp_W0 = (const float*)d_in[13];
    const float* exp_b0 = (const float*)d_in[14];
    const float* exp_W1 = (const float*)d_in[15];
    const float* exp_b1 = (const float*)d_in[16];
    const float* exp_W2 = (const float*)d_in[17];
    const float* exp_b2 = (const float*)d_in[18];
    const float* exp_Wo = (const float*)d_in[19];
    const float* exp_bo = (const float*)d_in[20];

    _Float16* wsf = (_Float16*)d_ws;
    float*    outp = (float*)d_out;

    // Pre-swizzle all matmul weights into WMMA B-fragment layout (f16).
    swizzle_w<<<64, 256, 0, stream>>>(enc_W1, wsf + OFF_ENC1,  52, 128, 2, 8, 1);
    swizzle_w<<<64, 256, 0, stream>>>(enc_W2, wsf + OFF_ENC2, 128, 128, 4, 8, 1);
    swizzle_w<<<16, 256, 0, stream>>>(pol_W0, wsf + OFF_POL0,   5,  64, 1, 4, 1);
    swizzle_w<<<16, 256, 0, stream>>>(pol_W1, wsf + OFF_POL1,  64,  64, 2, 4, 1);
    swizzle_w<<<16, 256, 0, stream>>>(pol_W2, wsf + OFF_POL2,  64,  64, 2, 4, 1);
    swizzle_w<<<256, 256, 0, stream>>>(exp_W0, wsf + OFF_EW0, 193, 128, 6, 8, 7);
    swizzle_w<<<256, 256, 0, stream>>>(exp_W1, wsf + OFF_EW1, 128, 128, 4, 8, 7);
    swizzle_w<<<256, 256, 0, stream>>>(exp_W2, wsf + OFF_EW2, 128, 128, 4, 8, 7);

    moe_inr_main<<<N_PTS / ROWS_PER_BLK, THREADS, 0, stream>>>(
        x, wsf,
        enc_b1, enc_b2,
        pol_b0, pol_b1, pol_b2,
        pol_Wl, pol_bl,
        exp_b0, exp_b1, exp_b2,
        exp_Wo, exp_bo,
        outp);
}
